// KMEAttention_48627619726105
// MI455X (gfx1250) — matrix operand order repaired and checked by reading, not compiled, not run on hardware
//
#include <hip/hip_runtime.h>

// ---------------------------------------------------------------------------
// KME attention pipeline for MI455X (gfx1250, wave32, WMMA 16x16x32 bf16,
// TDM tensor_load_to_lds staging, pure-b128 fragment loads, pre-converted
// bf16 weights so no fp32->bf16 conversion remains in the hot GEMM loops)
// ---------------------------------------------------------------------------

#define DEV __device__ __forceinline__

typedef __attribute__((ext_vector_type(16))) __bf16 v16bf;
typedef __attribute__((ext_vector_type(8)))  __bf16 v8bf;
typedef __attribute__((ext_vector_type(8)))  float  v8f;
typedef __attribute__((ext_vector_type(4)))  unsigned int v4ui;
typedef __attribute__((ext_vector_type(8)))  int    v8i;
typedef __attribute__((ext_vector_type(4)))  int    v4i;

constexpr int B = 4, S = 1024, M = 8, D = 64, H = 8, NF = 32, HD = H * D;
constexpr float RFF_SCALE  = 0.17677669529663687f;  // 1/sqrt(NF)
constexpr float ATTN_SCALE = 0.125f;                // 1/sqrt(D)

DEV v8f wmma_bf16(v16bf a, v16bf b, v8f c) {
  // D = A(16x32) * B(32x16) + C, fp32 accumulate
  return __builtin_amdgcn_wmma_f32_16x16x32_bf16(false, a, false, b,
                                                 (short)0, c, false, false);
}

// A fragment: 16 x 32 chunk of a row-major bf16 tile (leading dim ld).
// ISA layout: lane 0-15 row=lane, K={0..7, 16..23}; lanes 16-31: K += 8.
// Each lane's data is two contiguous 16-byte runs -> two b128 loads.
DEV v16bf frag_a(const __bf16* t, int ld, int lane, int kbase) {
  const int m = lane & 15, half = (lane >> 4) & 1;
  const __bf16* p = t + m * ld + kbase + 8 * half;
  const v8bf lo = *(const v8bf*)p;         // K = kbase+8h .. +7
  const v8bf hi = *(const v8bf*)(p + 16);  // K = kbase+16+8h .. +7
  v16bf a;
#pragma unroll
  for (int e = 0; e < 8; ++e) { a[e] = lo[e]; a[8 + e] = hi[e]; }
  return a;
}

// B fragment from K-contiguous source: element (k, n) at t[n*ld + k].
// Lane = col n; lanes 0-15 K={0..7, 16..23}; lanes 16-31: K += 8.
// Each lane's data is two contiguous 16-byte runs -> two b128 loads.
DEV v16bf frag_bT(const __bf16* t, int ld, int lane, int kbase) {
  const int n = lane & 15, half = (lane >> 4) & 1;
  const __bf16* p = t + n * ld + kbase + 8 * half;
  const v8bf lo = *(const v8bf*)p;
  const v8bf hi = *(const v8bf*)(p + 16);
  v16bf b;
#pragma unroll
  for (int e = 0; e < 8; ++e) { b[e] = lo[e]; b[8 + e] = hi[e]; }
  return b;
}

// ---------------------------------------------------------------------------
// Kernel 0: one-shot weight conversion to bf16 (+ freqs transpose to [H,NF,D])
// ---------------------------------------------------------------------------
__global__ __launch_bounds__(256) void k_cvt(const float* __restrict__ Wq,
                                             const float* __restrict__ Wk,
                                             const float* __restrict__ Wv,
                                             const float* __restrict__ Wo,
                                             const float* __restrict__ freqs,
                                             __bf16* __restrict__ Wq_b,
                                             __bf16* __restrict__ Wk_b,
                                             __bf16* __restrict__ Wv_b,
                                             __bf16* __restrict__ Wo_b,
                                             __bf16* __restrict__ freqsT) {
  const int i = blockIdx.x * 256 + threadIdx.x;
  constexpr int NW = HD * D;  // 32768 (all Linear weights are this size)
  if (i < NW) {
    Wq_b[i] = (__bf16)Wq[i];
    Wk_b[i] = (__bf16)Wk[i];
    Wv_b[i] = (__bf16)Wv[i];
    Wo_b[i] = (__bf16)Wo[i];
  }
  if (i < H * D * NF) {  // 16384: [H,D,NF] -> [H,NF,D]
    const int h = i / (D * NF), rem = i % (D * NF);
    const int d = rem / NF, f = rem % NF;
    freqsT[(h * NF + f) * D + d] = (__bf16)freqs[i];
  }
}

// ---------------------------------------------------------------------------
// Kernel 1: softmax over M of q/k log-weights
// ---------------------------------------------------------------------------
__global__ __launch_bounds__(256) void k_wsoftmax(const float* __restrict__ qlw,
                                                  const float* __restrict__ klw,
                                                  float* __restrict__ qw,
                                                  float* __restrict__ kw) {
  const int i = blockIdx.x * 256 + threadIdx.x;
  if (i >= B * S) return;
#pragma unroll
  for (int j = 0; j < 2; ++j) {
    const float* src = (j == 0 ? qlw : klw) + i * M;
    float* dst = (j == 0 ? qw : kw) + i * M;
    float mx = src[0];
#pragma unroll
    for (int m = 1; m < M; ++m) mx = fmaxf(mx, src[m]);
    float e[M], sm = 0.f;
#pragma unroll
    for (int m = 0; m < M; ++m) { e[m] = __expf(src[m] - mx); sm += e[m]; }
    const float inv = 1.f / sm;
#pragma unroll
    for (int m = 0; m < M; ++m) dst[m] = e[m] * inv;
  }
}

// ---------------------------------------------------------------------------
// Kernel 2: fused projection + RFF encode + weighted atom reduction.
// Workgroup = (b, 2 s-positions) -> 16 rows of (s,m). Wave w == head h.
// enc[b,h,s,:] = sum_m w[b,s,m] * [cos(qa@F), sin(qa@F)] * rff_scale
// ---------------------------------------------------------------------------
__global__ __launch_bounds__(256) void k_encode(const float* __restrict__ atoms,
                                                const float* __restrict__ wsm,
                                                const __bf16* __restrict__ Wb,
                                                const __bf16* __restrict__ freqsT,
                                                __bf16* __restrict__ enc) {
  __shared__ __bf16 atile[16 * 64];   // (s,m) rows x D, bf16
  __shared__ __bf16 qa[8][16 * 64];   // per-head projected tile
  __shared__ float wrow[16];
  const int blk = blockIdx.x;
  const int b = blk >> 9;            // S/2 = 512 tiles per batch
  const int s0 = (blk & 511) * 2;
  const int tid = threadIdx.x, lane = tid & 31, h = tid >> 5;
  const int half = (lane >> 4) & 1, n16 = lane & 15;

  for (int i = tid; i < 16 * 64; i += 256) {
    const int row = i >> 6, d = i & 63;
    atile[i] = (__bf16)atoms[((b * S + s0 + (row >> 3)) * M + (row & 7)) * D + d];
  }
  if (tid < 16)
    wrow[tid] = wsm[(b * S + s0 + (tid >> 3)) * M + (tid & 7)];
  __syncthreads();

  const v16bf a0 = frag_a(atile, 64, lane, 0);
  const v16bf a1 = frag_a(atile, 64, lane, 32);

  // projection: 16x64 <- 16x64 @ (64 x 64) slice of W for head h
  // W is [HD, D] row-major bf16: element (k, n) = Wb[(n0+n)*D + k] -> frag_bT
#pragma unroll
  for (int t = 0; t < 4; ++t) {
    const __bf16* wp = Wb + (h * 64 + t * 16) * D;
    v8f c = {};
    c = wmma_bf16(a0, frag_bT(wp, D, lane, 0), c);
    c = wmma_bf16(a1, frag_bT(wp, D, lane, 32), c);
#pragma unroll
    for (int r = 0; r < 8; ++r)
      qa[h][(r + 8 * half) * 64 + t * 16 + n16] = (__bf16)c[r];
  }
  __syncthreads();

  // RFF: p = qa @ freqs[h] (16x32), then phi = [cos,sin]; reduce over m.
  // freqsT is [H,NF,D]: element (k, n) = freqsT[(h*NF + f0+n)*D + k] -> frag_bT
  const v16bf ra0 = frag_a(qa[h], 64, lane, 0);
  const v16bf ra1 = frag_a(qa[h], 64, lane, 32);
#pragma unroll
  for (int f = 0; f < 2; ++f) {
    const __bf16* fp = freqsT + (h * NF + f * 16) * D;
    v8f p = {};
    p = wmma_bf16(ra0, frag_bT(fp, D, lane, 0), p);
    p = wmma_bf16(ra1, frag_bT(fp, D, lane, 32), p);
    // rows r=0..7 of this lane's accumulator are the 8 atoms of s = s0+half
    float ca = 0.f, sa = 0.f;
#pragma unroll
    for (int r = 0; r < 8; ++r) {
      const float w = wrow[r + 8 * half];
      ca += w * __cosf(p[r]);
      sa += w * __sinf(p[r]);
    }
    const int s = s0 + half, col = f * 16 + n16;
    __bf16* dst = enc + ((b * H + h) * S + s) * D + col;
    dst[0]  = (__bf16)(ca * RFF_SCALE);
    dst[32] = (__bf16)(sa * RFF_SCALE);
  }
}

// ---------------------------------------------------------------------------
// Kernel 3: value projection -> vat[B][M][H][D][S] bf16 (K-major so the PV
// GEMM's B-fragments are two contiguous 16B runs per lane -> b128 loads)
// ---------------------------------------------------------------------------
__global__ __launch_bounds__(256) void k_vproj(const float* __restrict__ atoms,
                                               const __bf16* __restrict__ Wb,
                                               __bf16* __restrict__ vat) {
  __shared__ __bf16 atile[16 * 64];
  const int blk = blockIdx.x;
  const int b = blk >> 9;
  const int s0 = (blk & 511) * 2;
  const int tid = threadIdx.x, lane = tid & 31, h = tid >> 5;
  const int half = (lane >> 4) & 1, n16 = lane & 15;

  for (int i = tid; i < 16 * 64; i += 256) {
    const int row = i >> 6, d = i & 63;
    atile[i] = (__bf16)atoms[((b * S + s0 + (row >> 3)) * M + (row & 7)) * D + d];
  }
  __syncthreads();

  const v16bf a0 = frag_a(atile, 64, lane, 0);
  const v16bf a1 = frag_a(atile, 64, lane, 32);
#pragma unroll
  for (int t = 0; t < 4; ++t) {
    const __bf16* wp = Wb + (h * 64 + t * 16) * D;
    v8f c = {};
    c = wmma_bf16(a0, frag_bT(wp, D, lane, 0), c);
    c = wmma_bf16(a1, frag_bT(wp, D, lane, 32), c);
    const int d = t * 16 + n16;
#pragma unroll
    for (int r = 0; r < 8; ++r) {
      const int row = r + 8 * half;
      const int s = s0 + (row >> 3), m = row & 7;
      vat[(size_t)(((b * M + m) * H + h) * D + d) * S + s] = (__bf16)c[r];
    }
  }
}

// ---------------------------------------------------------------------------
// Kernel 4: fused scores + softmax + (attn @ V).  Workgroup = (b, h, 16 q).
// Score stripe [16 x 1024] lives in LDS only (never touches HBM).
// Q tile staged by the Tensor Data Mover (TDM).
// Wave w handles: phase1 -> 8 key tiles; phase3 -> atom m = w.
// ---------------------------------------------------------------------------
__global__ __launch_bounds__(256) void k_attn(const __bf16* __restrict__ qenc,
                                              const __bf16* __restrict__ kenc,
                                              const __bf16* __restrict__ vat,
                                              __bf16* __restrict__ outhd) {
  // one shared block so qtile sits at LDS offset 0 (TDM descriptor lds_addr)
  __shared__ __align__(16) unsigned char smem[(16 * 64 + 16 * 1024) * 2];
  __bf16* qtile = (__bf16*)smem;               //  2 KB
  __bf16* sc = (__bf16*)(smem + 16 * 64 * 2);  // 32 KB score/prob stripe

  const int blk = blockIdx.x;
  const int qt = blk & 63, h = (blk >> 6) & 7, b = blk >> 9;
  const int q0 = qt * 16;
  const int tid = threadIdx.x, lane = tid & 31, wv = tid >> 5;
  const int half = (lane >> 4) & 1, n16 = lane & 15;

  // ---- stage Q tile (16x64 contiguous in memory = 2 KB linear tile) ----
#if defined(__gfx1250__) && __has_builtin(__builtin_amdgcn_tensor_load_to_lds)
  if (wv == 0) {
    const unsigned long long ga =
        (unsigned long long)(const void*)(qenc + ((b * H + h) * S + q0) * D);
    v4ui g0;
    g0[0] = 1u;                                                // count=1
    g0[1] = 0u;                                                // lds_addr = 0
    g0[2] = (unsigned)ga;                                      // global lo
    g0[3] = (unsigned)((ga >> 32) & 0x1FFFFFFu) | (2u << 30);  // hi | type=2
    v8i g1;
    g1[0] = 0x00010000;                      // data_size=2B, no mask/flags
    g1[1] = (int)((1024u & 0xFFFFu) << 16);  // tensor_dim0 = 1024 (lo16)
    g1[2] = 0x00010000;                      // td0 hi | tensor_dim1 = 1
    g1[3] = (int)(1024u << 16);              // tile_dim0 = 1024
    g1[4] = 1;                               // tile_dim1 = 1
    g1[5] = 1024;                            // tensor_dim0_stride = 1024
    g1[6] = 0;
    g1[7] = 0;
    v4i z4 = {0, 0, 0, 0};
    v8i z8 = {0, 0, 0, 0, 0, 0, 0, 0};
    __builtin_amdgcn_tensor_load_to_lds(g0, g1, z4, z4, z8, 0);
    __builtin_amdgcn_s_wait_tensorcnt(0);
  }
#else
  for (int i = tid; i < 16 * 64; i += 256) {
    const int row = i >> 6, d = i & 63;
    qtile[i] = qenc[((b * H + h) * S + q0 + row) * D + d];
  }
#endif
  __syncthreads();

  // phase 1: S_tile = Q (16x64) @ K^T, scaled, stored bf16 in LDS
  const v16bf a0 = frag_a(qtile, 64, lane, 0);
  const v16bf a1 = frag_a(qtile, 64, lane, 32);
  const __bf16* kbase = kenc + (b * H + h) * S * D;
#pragma unroll 2
  for (int j = 0; j < 8; ++j) {
    const int kc = (wv * 8 + j) * 16;
    v8f c = {};
    c = wmma_bf16(a0, frag_bT(kbase + kc * D, D, lane, 0), c);
    c = wmma_bf16(a1, frag_bT(kbase + kc * D, D, lane, 32), c);
#pragma unroll
    for (int r = 0; r < 8; ++r)
      sc[(r + 8 * half) * 1024 + kc + n16] = (__bf16)(c[r] * ATTN_SCALE);
  }
  __syncthreads();

  // phase 2: row softmax over 1024 keys (16 threads per row, shfl width 16)
  {
    const int g = tid >> 4, gi = tid & 15;
    float mx = -3.0e38f;
    for (int j = 0; j < 64; ++j)
      mx = fmaxf(mx, (float)sc[g * 1024 + gi + j * 16]);
    for (int o = 8; o; o >>= 1) mx = fmaxf(mx, __shfl_xor(mx, o, 16));
    float sm = 0.f;
    for (int j = 0; j < 64; ++j)
      sm += __expf((float)sc[g * 1024 + gi + j * 16] - mx);
    for (int o = 8; o; o >>= 1) sm += __shfl_xor(sm, o, 16);
    const float inv = 1.f / sm;
    for (int j = 0; j < 64; ++j) {
      const int idx = g * 1024 + gi + j * 16;
      sc[idx] = (__bf16)(__expf((float)sc[idx] - mx) * inv);
    }
  }
  __syncthreads();

  // phase 3: out[16 x 64] = P (16x1024) @ V_m ; V is K-major -> b128 loads
  const int m = wv;
  const __bf16* vbase = vat + (size_t)(((b * M + m) * H + h) * D) * S;
  v8f acc[4] = {};
#pragma unroll 2
  for (int kk = 0; kk < 32; ++kk) {
    const v16bf ap = frag_a(sc, 1024, lane, kk * 32);
    if (kk < 31)
      __builtin_prefetch(vbase + n16 * S + (kk + 1) * 32, 0, 0);
#pragma unroll
    for (int t = 0; t < 4; ++t)
      acc[t] = wmma_bf16(ap, frag_bT(vbase + (size_t)(t * 16) * S, S, lane,
                                     kk * 32),
                         acc[t]);
  }
#pragma unroll
  for (int t = 0; t < 4; ++t)
#pragma unroll
    for (int r = 0; r < 8; ++r) {
      const int row = r + 8 * half;
      outhd[(((b * S + q0 + row) * M + m) * H + h) * D + t * 16 + n16] =
          (__bf16)acc[t][r];
    }
}

// ---------------------------------------------------------------------------
// Kernel 5: Wo projection -> out_atoms (fp32), then mean over m + Ww ->
// new_log_weights.  Workgroup = (b, 2 s).  4 waves, wave = n-tile of D.
// ---------------------------------------------------------------------------
__global__ __launch_bounds__(128) void k_out(const __bf16* __restrict__ outhd,
                                             const __bf16* __restrict__ Wo_b,
                                             const float* __restrict__ Ww,
                                             const float* __restrict__ qlw,
                                             float* __restrict__ out_atoms,
                                             float* __restrict__ nlw) {
  __shared__ float oat[16 * 64];
  const int blk = blockIdx.x;
  const int b = blk >> 9;
  const int s0 = (blk & 511) * 2;
  const int tid = threadIdx.x, lane = tid & 31, t = tid >> 5;
  const int half = (lane >> 4) & 1, n16 = lane & 15;

  // A rows are (s,m) pairs; row stride in outhd is exactly HD.
  // Wo is [D, HD] row-major bf16: element (k, n) = Wo_b[(t*16+n)*HD + k]
  const __bf16* abase = outhd + (size_t)(b * S + s0) * M * HD;
  const __bf16* wp = Wo_b + (t * 16) * HD;
  v8f c = {};
  for (int kk = 0; kk < 16; ++kk) {
    const v16bf af = frag_a(abase, HD, lane, kk * 32);
    c = wmma_bf16(af, frag_bT(wp, HD, lane, kk * 32), c);
  }
#pragma unroll
  for (int r = 0; r < 8; ++r) {
    const int row = r + 8 * half;
    const float v = c[r];
    out_atoms[((b * S + s0 + (row >> 3)) * M + (row & 7)) * D + t * 16 + n16] = v;
    oat[row * 64 + t * 16 + n16] = v;
  }
  __syncthreads();

  if (tid < 16) {
    const int sl = tid >> 3, m = tid & 7, s = s0 + sl;
    float acc = qlw[(b * S + s) * M + m];
    for (int d = 0; d < D; ++d) {
      float f = 0.f;
#pragma unroll
      for (int mm = 0; mm < 8; ++mm) f += oat[(sl * 8 + mm) * 64 + d];
      acc += 0.125f * f * Ww[m * D + d];
    }
    nlw[(b * S + s) * M + m] = acc;
  }
}

// ---------------------------------------------------------------------------
extern "C" void kernel_launch(void* const* d_in, const int* in_sizes, int n_in,
                              void* d_out, int out_size, void* d_ws,
                              size_t ws_size, hipStream_t stream) {
  (void)in_sizes; (void)n_in; (void)out_size; (void)ws_size;
  const float* q_atoms = (const float*)d_in[0];
  const float* q_logw  = (const float*)d_in[1];
  const float* k_atoms = (const float*)d_in[2];
  const float* k_logw  = (const float*)d_in[3];
  const float* v_atoms = (const float*)d_in[4];
  // d_in[5] (v_logw) is unused by the reference
  const float* Wq    = (const float*)d_in[6];
  const float* Wk    = (const float*)d_in[7];
  const float* Wv    = (const float*)d_in[8];
  const float* Wo    = (const float*)d_in[9];
  const float* Ww    = (const float*)d_in[10];
  const float* freqs = (const float*)d_in[11];

  // workspace carve-up (~72.6 MB total)
  char* ws = (char*)d_ws;
  size_t off = 0;
  auto carve = [&](size_t bytes) {
    char* p = ws + off;
    off += (bytes + 255) & ~(size_t)255;
    return p;
  };
  float*  qw     = (float*)carve(sizeof(float) * B * S * M);           // 128 KB
  float*  kw     = (float*)carve(sizeof(float) * B * S * M);           // 128 KB
  __bf16* q_enc  = (__bf16*)carve(sizeof(__bf16) * B * H * S * D);     //   4 MB
  __bf16* k_enc  = (__bf16*)carve(sizeof(__bf16) * B * H * S * D);     //   4 MB
  __bf16* vat    = (__bf16*)carve(sizeof(__bf16) * B * S * M * H * D); //  32 MB
  __bf16* outhd  = (__bf16*)carve(sizeof(__bf16) * B * S * M * H * D); //  32 MB
  __bf16* Wq_b   = (__bf16*)carve(sizeof(__bf16) * HD * D);            //  64 KB
  __bf16* Wk_b   = (__bf16*)carve(sizeof(__bf16) * HD * D);            //  64 KB
  __bf16* Wv_b   = (__bf16*)carve(sizeof(__bf16) * HD * D);            //  64 KB
  __bf16* Wo_b   = (__bf16*)carve(sizeof(__bf16) * D * HD);            //  64 KB
  __bf16* freqsT = (__bf16*)carve(sizeof(__bf16) * H * NF * D);        //  32 KB

  float* out_atoms = (float*)d_out;              // [B,S,M,D]
  float* nlw       = out_atoms + B * S * M * D;  // [B,S,M]

  k_cvt<<<dim3((HD * D + 255) / 256), dim3(256), 0, stream>>>(
      Wq, Wk, Wv, Wo, freqs, Wq_b, Wk_b, Wv_b, Wo_b, freqsT);
  k_wsoftmax<<<dim3((B * S + 255) / 256), dim3(256), 0, stream>>>(
      q_logw, k_logw, qw, kw);
  k_encode<<<dim3(B * S / 2), dim3(256), 0, stream>>>(q_atoms, qw, Wq_b,
                                                      freqsT, q_enc);
  k_encode<<<dim3(B * S / 2), dim3(256), 0, stream>>>(k_atoms, kw, Wk_b,
                                                      freqsT, k_enc);
  k_vproj<<<dim3(B * S / 2), dim3(256), 0, stream>>>(v_atoms, Wv_b, vat);
  k_attn<<<dim3(B * H * (S / 16)), dim3(256), 0, stream>>>(q_enc, k_enc, vat,
                                                           outhd);
  k_out<<<dim3(B * S / 2), dim3(128), 0, stream>>>(outhd, Wo_b, Ww, q_logw,
                                                   out_atoms, nlw);
}
